// EnhancedTernaryLinear_32968168964710
// MI455X (gfx1250) — matrix operand order repaired and checked by reading, not compiled, not run on hardware
//
#include <hip/hip_runtime.h>
#include <hip/hip_bf16.h>

// EnhancedTernaryLinear: out[M,N] = (x[M,K] @ W[N,K]^T) * scale[N] + bias[N]
// M = 8192, K = N = 4096, W ternary int8.
// Compute-bound (AI ~1000 FLOP/B) -> bf16 WMMA 16x16x32 with f32 accum.
// W (and x) are pre-converted to bf16 once in d_ws so the GEMM hot loop is
// pure b128 copy + WMMA (no per-tile int8->bf16 VALU redone 64x per W tile).

#define K_DIM 4096
#define N_DIM 4096
#define BM 128
#define BN 128
#define BK 32
#define LDSB (BK + 8) // bf16 elems per LDS row: 80 B (16B aligned, bank-spread)

typedef __bf16 v16bf __attribute__((ext_vector_type(16)));
typedef __bf16 v8bf  __attribute__((ext_vector_type(8)));
typedef float  v8f   __attribute__((ext_vector_type(8)));
typedef float  v4f   __attribute__((ext_vector_type(4)));

// A-matrix 16x32 bf16 fragment (ISA 7.12.2):
// lanes 0-15 (M=lane): VGPR0-3 = K 0..7,  VGPR4-7 = K 16..23
// lanes 16-31        : VGPR0-3 = K 8..15, VGPR4-7 = K 24..31
__device__ __forceinline__ v16bf frag_ld_A(const __bf16* rowbase, int half) {
  const v8bf lo = *(const v8bf*)(rowbase + half * 8);
  const v8bf hi = *(const v8bf*)(rowbase + 16 + half * 8);
  return __builtin_shufflevector(lo, hi, 0, 1, 2, 3, 4, 5, 6, 7,
                                 8, 9, 10, 11, 12, 13, 14, 15);
}

// B-matrix 32x16 bf16 fragment (same K order as A): lanes 0-15 (N=lane) hold
// K 0..7, 16..23; lanes 16-31 hold K 8..15, 24..31.
__device__ __forceinline__ v16bf frag_ld_B(const __bf16* rowbase, int half) {
  const v8bf lo = *(const v8bf*)(rowbase + half * 8);
  const v8bf hi = *(const v8bf*)(rowbase + 16 + half * 8);
  return __builtin_shufflevector(lo, hi, 0, 1, 2, 3, 4, 5, 6, 7,
                                 8, 9, 10, 11, 12, 13, 14, 15);
}

// ---- pre-conversion kernels (memory-bound, run once per call) -------------

// x: fp32 -> bf16, 8 elems/thread
__global__ __launch_bounds__(256) void conv_x_bf16(const float* __restrict__ in,
                                                   __bf16* __restrict__ out) {
  const size_t i = ((size_t)blockIdx.x * blockDim.x + threadIdx.x) * 8;
  const v4f a = *(const v4f*)(in + i);
  const v4f b = *(const v4f*)(in + i + 4);
  v8bf o;
#pragma unroll
  for (int j = 0; j < 4; ++j) { o[j] = (__bf16)a[j]; o[4 + j] = (__bf16)b[j]; }
  *(v8bf*)(out + i) = o;
}

// W: ternary int8 -> bf16, 16 elems/thread
__global__ __launch_bounds__(256) void conv_w_bf16(
    const signed char* __restrict__ in, __bf16* __restrict__ out) {
  const size_t i = ((size_t)blockIdx.x * blockDim.x + threadIdx.x) * 16;
  const int4 t = *(const int4*)(in + i);
  int wv[4] = {t.x, t.y, t.z, t.w};
  const signed char* c = (const signed char*)wv;
  v8bf q0, q1;
#pragma unroll
  for (int j = 0; j < 8; ++j) {
    q0[j] = (__bf16)(float)c[j];
    q1[j] = (__bf16)(float)c[8 + j];
  }
  *(v8bf*)(out + i)     = q0;
  *(v8bf*)(out + i + 8) = q1;
}

// ---- GEMM -----------------------------------------------------------------

template <bool PA, bool PB>
__global__ __launch_bounds__(256) void ternary_linear_wmma(
    const float* __restrict__ xf, const __bf16* __restrict__ xb,
    const signed char* __restrict__ wq, const __bf16* __restrict__ wb,
    const float* __restrict__ scale, const float* __restrict__ bias,
    float* __restrict__ out) {
  __shared__ __attribute__((aligned(16))) __bf16 sA[2][BM * LDSB];
  __shared__ __attribute__((aligned(16))) __bf16 sB[2][BN * LDSB];

  const int tid  = threadIdx.x;
  const int lane = tid & 31;
  const int wave = tid >> 5;
  const int waveM = (wave & 3) * 32;  // 4 waves tile M
  const int waveN = (wave >> 2) * 64; // 2 waves tile N
  const int laneRow = lane & 15;
  const int half    = lane >> 4;

  // Global staging: each thread owns one (row, 16-wide K segment).
  const int ldRow = tid >> 1;
  const int ldK   = (tid & 1) * 16;
  const size_t gmRow = (size_t)(blockIdx.y * BM + ldRow) * K_DIM;
  const size_t gnRow = (size_t)(blockIdx.x * BN + ldRow) * K_DIM;

  v4f  xr[4];  // raw path: 16 floats of x
  v8bf xbr[2]; // preconverted path: 16 bf16 of x
  int  wv[4];  // raw path: 16 int8 of W
  v8bf wbr[2]; // preconverted path: 16 bf16 of W

  auto g_load = [&](int kc) {
    if constexpr (PA) {
      const v8bf* pa = (const v8bf*)(xb + gmRow + kc + ldK);
      xbr[0] = pa[0]; xbr[1] = pa[1];
    } else {
      const v4f* pa = (const v4f*)(xf + gmRow + kc + ldK);
      xr[0] = pa[0]; xr[1] = pa[1]; xr[2] = pa[2]; xr[3] = pa[3];
    }
    if constexpr (PB) {
      const v8bf* pb = (const v8bf*)(wb + gnRow + kc + ldK);
      wbr[0] = pb[0]; wbr[1] = pb[1];
    } else {
      const int4 t = *(const int4*)(wq + gnRow + kc + ldK);
      wv[0] = t.x; wv[1] = t.y; wv[2] = t.z; wv[3] = t.w;
    }
  };

  auto l_store = [&](int buf) {
    __bf16* da = &sA[buf][ldRow * LDSB + ldK];
    if constexpr (PA) {
      *(v8bf*)(da)     = xbr[0];
      *(v8bf*)(da + 8) = xbr[1];
    } else {
      v8bf p0, p1;
#pragma unroll
      for (int i = 0; i < 8; ++i) {
        p0[i] = (__bf16)xr[i >> 2][i & 3];
        p1[i] = (__bf16)xr[2 + (i >> 2)][i & 3];
      }
      *(v8bf*)(da)     = p0;
      *(v8bf*)(da + 8) = p1;
    }
    __bf16* db = &sB[buf][ldRow * LDSB + ldK];
    if constexpr (PB) {
      *(v8bf*)(db)     = wbr[0];
      *(v8bf*)(db + 8) = wbr[1];
    } else {
      const signed char* c = (const signed char*)wv;
      v8bf q0, q1;
#pragma unroll
      for (int i = 0; i < 8; ++i) {
        q0[i] = (__bf16)(float)c[i];
        q1[i] = (__bf16)(float)c[8 + i];
      }
      *(v8bf*)(db)     = q0;
      *(v8bf*)(db + 8) = q1;
    }
  };

  v8f acc[2][4] = {};
  v16bf afrag[2], bfrag[4];

  auto compute = [&](int buf) {
#pragma unroll
    for (int mt = 0; mt < 2; ++mt)
      afrag[mt] = frag_ld_A(&sA[buf][(waveM + mt * 16 + laneRow) * LDSB], half);
#pragma unroll
    for (int nt = 0; nt < 4; ++nt)
      bfrag[nt] = frag_ld_B(&sB[buf][(waveN + nt * 16 + laneRow) * LDSB], half);
#pragma unroll
    for (int mt = 0; mt < 2; ++mt)
#pragma unroll
      for (int nt = 0; nt < 4; ++nt)
        acc[mt][nt] = __builtin_amdgcn_wmma_f32_16x16x32_bf16(
            false, afrag[mt], false, bfrag[nt], (short)0, acc[mt][nt], false,
            false);
  };

  constexpr int KT = K_DIM / BK; // 128 K-steps

  g_load(0);
  l_store(0);
  __syncthreads();

  int buf = 0;
  for (int kt = 0; kt < KT; ++kt) {
    if (kt + 2 < KT) { // keep HBM ahead of the pipeline
      if constexpr (PA)
        __builtin_prefetch(xb + gmRow + (kt + 2) * BK + ldK, 0, 3);
      else
        __builtin_prefetch(xf + gmRow + (kt + 2) * BK + ldK, 0, 3);
      if constexpr (PB)
        __builtin_prefetch(wb + gnRow + (kt + 2) * BK + ldK, 0, 3);
      else
        __builtin_prefetch(wq + gnRow + (kt + 2) * BK + ldK, 0, 3);
    }
    if (kt + 1 < KT) g_load((kt + 1) * BK); // overlap global with WMMA
    compute(buf);
    if (kt + 1 < KT) l_store(buf ^ 1);
    __syncthreads();
    buf ^= 1;
  }

  // Epilogue: C/D layout (ISA 7.12.2): VGPR r -> M = r + 8*half, lane&15 -> N.
#pragma unroll
  for (int nt = 0; nt < 4; ++nt) {
    const int gn = blockIdx.x * BN + waveN + nt * 16 + laneRow;
    const float s = scale[gn];
    const float b = bias[gn];
#pragma unroll
    for (int mt = 0; mt < 2; ++mt) {
      const int gmBase = blockIdx.y * BM + waveM + mt * 16 + half * 8;
#pragma unroll
      for (int r = 0; r < 8; ++r) {
        out[(size_t)(gmBase + r) * N_DIM + gn] = acc[mt][nt][r] * s + b;
      }
    }
  }
}

extern "C" void kernel_launch(void* const* d_in, const int* in_sizes, int n_in,
                              void* d_out, int out_size, void* d_ws,
                              size_t ws_size, hipStream_t stream) {
  (void)n_in; (void)out_size;
  const float*       x  = (const float*)d_in[0];
  const signed char* w  = (const signed char*)d_in[1];
  const float*       sc = (const float*)d_in[2];
  const float*       bi = (const float*)d_in[3];
  float*             o  = (float*)d_out;

  const int    M     = in_sizes[0] / K_DIM; // 8192
  const size_t nXel  = (size_t)M * K_DIM;
  const size_t nWel  = (size_t)N_DIM * K_DIM;
  const size_t needB = nWel * sizeof(__bf16);         // 32 MB
  const size_t needA = nXel * sizeof(__bf16);         // 64 MB

  dim3 grid(N_DIM / BN, M / BM); // (32, 64)
  dim3 block(256);

  const bool preB = ws_size >= needB;
  const bool preA = ws_size >= needA + needB;

  __bf16* wb = (__bf16*)d_ws;                       // [0, 32MB)
  __bf16* xb = (__bf16*)((char*)d_ws + needB);      // [32MB, 96MB)

  if (preB)
    conv_w_bf16<<<dim3((unsigned)(nWel / (256 * 16))), block, 0, stream>>>(w, wb);
  if (preA)
    conv_x_bf16<<<dim3((unsigned)(nXel / (256 * 8))), block, 0, stream>>>(x, xb);

  if (preA && preB)
    ternary_linear_wmma<true, true>
        <<<grid, block, 0, stream>>>(x, xb, w, wb, sc, bi, o);
  else if (preB)
    ternary_linear_wmma<false, true>
        <<<grid, block, 0, stream>>>(x, xb, w, wb, sc, bi, o);
  else
    ternary_linear_wmma<false, false>
        <<<grid, block, 0, stream>>>(x, xb, w, wb, sc, bi, o);
}
